// MultiHeadSelfAttention_32066225832666
// MI455X (gfx1250) — compile-verified, matrix operand order checked by reading, not run on hardware
//
#include <hip/hip_runtime.h>

// ---- problem constants (match reference) ----
#define BATCH 2
#define SEQ   2048
#define EMB   1024
#define HEADS 16
#define HD    64              // head dim
#define SOFTMAX_SCALE 0.125f  // 1/sqrt(64)

typedef __attribute__((ext_vector_type(16))) __bf16          v16bf;
typedef __attribute__((ext_vector_type(8)))  float           v8f;
typedef __attribute__((ext_vector_type(8)))  unsigned short  v8us;
typedef __attribute__((ext_vector_type(16))) unsigned short  v16us;

// ---- helpers ----
static __device__ __forceinline__ unsigned short f2bf_u(float f) {
    union { float f; unsigned u; } v; v.f = f;
    unsigned r = v.u + 0x7fffu + ((v.u >> 16) & 1u);   // round-to-nearest-even
    return (unsigned short)(r >> 16);
}

// WMMA A/B operand loader from bf16 memory (global or LDS).
// base = &src[row*ld + k0]; lane row = lo, K-halves per ISA 7.12.2:
// lanes 0-15: K kb+0..7 / kb+16..23 with kb=0; lanes 16-31: kb=8.
// Both 8-element halves are 16-byte aligned contiguous runs -> b128 loads.
static __device__ __forceinline__ v16bf load_ab16(const unsigned short* base, int kb) {
    const v8us lo8 = *(const v8us*)(base + kb);
    const v8us hi8 = *(const v8us*)(base + kb + 16);
    const v16us c = __builtin_shufflevector(lo8, hi8,
        0, 1, 2, 3, 4, 5, 6, 7, 8, 9, 10, 11, 12, 13, 14, 15);
    return __builtin_bit_cast(v16bf, c);
}

static __device__ __forceinline__ v8f wmma_bf16(v16bf a, v16bf b, v8f c) {
    return __builtin_amdgcn_wmma_f32_16x16x32_bf16(
        /*neg_a=*/false, a, /*neg_b=*/false, b,
        /*c_mod=*/(short)0, c, /*reuse_a=*/false, /*reuse_b=*/false);
}

// =====================================================================
// Kernel 0: one-shot fp32 -> bf16 conversion of x and the four weights.
// Bandwidth-bound (~33 MB read); grid.y selects the array.
// =====================================================================
__global__ void cvt_bf16_kernel(const float* __restrict__ x,
                                const float* __restrict__ Wq,
                                const float* __restrict__ Wk,
                                const float* __restrict__ Wv,
                                const float* __restrict__ Wo,
                                unsigned short* __restrict__ xb,
                                unsigned short* __restrict__ Wqb,
                                unsigned short* __restrict__ Wkb,
                                unsigned short* __restrict__ Wvb,
                                unsigned short* __restrict__ Wob) {
    const int which = blockIdx.y;
    const float* s; unsigned short* d; int n;
    switch (which) {
        case 0:  s = x;  d = xb;  n = BATCH * SEQ * EMB; break;
        case 1:  s = Wq; d = Wqb; n = EMB * EMB; break;
        case 2:  s = Wk; d = Wkb; n = EMB * EMB; break;
        case 3:  s = Wv; d = Wvb; n = EMB * EMB; break;
        default: s = Wo; d = Wob; n = EMB * EMB; break;
    }
    const int stride = gridDim.x * blockDim.x * 8;
    for (int i = (blockIdx.x * blockDim.x + threadIdx.x) * 8; i < n; i += stride) {
        const float4 a = *(const float4*)(s + i);
        const float4 b = *(const float4*)(s + i + 4);
        v8us o;
        o[0] = f2bf_u(a.x); o[1] = f2bf_u(a.y); o[2] = f2bf_u(a.z); o[3] = f2bf_u(a.w);
        o[4] = f2bf_u(b.x); o[5] = f2bf_u(b.y); o[6] = f2bf_u(b.z); o[7] = f2bf_u(b.w);
        *(v8us*)(d + i) = o;
    }
}

// =====================================================================
// Kernel 1: fused QKV projection, bf16 inputs, wave tile 32x64 (2Mx4N).
// grid = (EMB/64, B*SEQ/32, 3), block = 32.
// Writes Q,K as bf16 [b,h,s,hd] and V transposed bf16 [b,h,hd,s].
// =====================================================================
__global__ void qkv_gemm_kernel(const unsigned short* __restrict__ xb,
                                const unsigned short* __restrict__ Wqb,
                                const unsigned short* __restrict__ Wkb,
                                const unsigned short* __restrict__ Wvb,
                                const float* __restrict__ qs,
                                const float* __restrict__ ks,
                                const float* __restrict__ vs,
                                unsigned short* __restrict__ Qb,
                                unsigned short* __restrict__ Kb,
                                unsigned short* __restrict__ Vtb) {
    const int lane = threadIdx.x & 31;
    const int lo = lane & 15;
    const int hi = lane >> 4;
    const int kb = hi * 8;

    const int nT0 = blockIdx.x * 4;      // first of 4 output-channel tiles
    const int mT0 = blockIdx.y * 2;      // first of 2 row tiles
    const int type = blockIdx.z;         // 0=Q 1=K 2=V

    const unsigned short* W  = (type == 0) ? Wqb : (type == 1) ? Wkb : Wvb;
    const float*          sc = (type == 0) ? qs  : (type == 1) ? ks  : vs;

    const unsigned short* aRow0 = xb + (size_t)(mT0 * 16 + lo) * EMB;
    const unsigned short* aRow1 = aRow0 + (size_t)16 * EMB;
    const unsigned short* bRow[4];
#pragma unroll
    for (int t = 0; t < 4; ++t)
        bRow[t] = W + (size_t)((nT0 + t) * 16 + lo) * EMB;

    v8f acc0[4] = {}, acc1[4] = {};
#pragma unroll 2
    for (int k0 = 0; k0 < EMB; k0 += 32) {
        const v16bf a0 = load_ab16(aRow0 + k0, kb);
        const v16bf a1 = load_ab16(aRow1 + k0, kb);
        __builtin_prefetch(aRow0 + k0 + 64, 0, 1);
#pragma unroll
        for (int t = 0; t < 4; ++t) {
            const v16bf b = load_ab16(bRow[t] + k0, kb);
            acc0[t] = wmma_bf16(a0, b, acc0[t]);
            acc1[t] = wmma_bf16(a1, b, acc1[t]);
        }
    }

#pragma unroll
    for (int t = 0; t < 4; ++t) {
        const int ng = (nT0 + t) * 16 + lo;     // output channel (fixed per lane)
        const float scv = sc[ng];
        const int h = ng >> 6, d = ng & 63;
#pragma unroll
        for (int m = 0; m < 2; ++m) {
            const v8f av = (m == 0 ? acc0[t] : acc1[t]) * scv;
#pragma unroll
            for (int r = 0; r < 8; ++r) {
                const int mg = (mT0 + m) * 16 + r + 8 * hi;   // C-layout row
                const int bb = mg >> 11, s = mg & (SEQ - 1);
                const unsigned short v = f2bf_u(av[r]);
                const size_t bh = (size_t)bb * HEADS + h;
                if (type == 2) {
                    Vtb[(bh * HD + d) * SEQ + s] = v;          // [b,h,d,s]
                } else {
                    unsigned short* dst = (type == 0) ? Qb : Kb;
                    dst[(bh * SEQ + s) * HD + d] = v;          // [b,h,s,d]
                }
            }
        }
    }
}

// =====================================================================
// Kernel 2: flash attention.  grid = (SEQ/16, HEADS, BATCH), block = 32.
// One wave owns a 16-row Q tile; streams keys in blocks of 32 with online
// softmax; P redistributed C-layout -> A-layout through LDS (b128 ds ops).
// =====================================================================
__global__ void attn_kernel(const unsigned short* __restrict__ Qb,
                            const unsigned short* __restrict__ Kb,
                            const unsigned short* __restrict__ Vtb,
                            unsigned short* __restrict__ Ctxb) {
    __shared__ __align__(16) unsigned short pb[16 * 32];   // P bounce (1 KB)

    const int lane = threadIdx.x & 31;
    const int lo = lane & 15;
    const int hi = lane >> 4;
    const int kb = hi * 8;

    const int qT = blockIdx.x;
    const int h  = blockIdx.y;
    const int bb = blockIdx.z;
    const size_t bh = (size_t)bb * HEADS + h;

    // Q tile, loaded once (Hd = 64 -> two A operands).
    const unsigned short* Qp = Qb + (bh * SEQ + (size_t)qT * 16) * HD;
    const v16bf aq0 = load_ab16(Qp + (size_t)lo * HD + 0,  kb);
    const v16bf aq1 = load_ab16(Qp + (size_t)lo * HD + 32, kb);

    const unsigned short* Kp = Kb  + bh * SEQ * HD;
    const unsigned short* Vp = Vtb + bh * HD * SEQ;

    v8f accO[4] = {};
    float mst[8], lst[8];
#pragma unroll
    for (int r = 0; r < 8; ++r) { mst[r] = -INFINITY; lst[r] = 0.f; }

    for (int j = 0; j < SEQ; j += 32) {
        // ---- scores: 2 column tiles of 16 keys ----
        v8f s0 = {}, s1 = {};
        {
            const unsigned short* k0p = Kp + (size_t)(j + lo) * HD;
            const unsigned short* k1p = Kp + (size_t)(j + 16 + lo) * HD;
            s0 = wmma_bf16(aq0, load_ab16(k0p + 0,  kb), s0);
            s0 = wmma_bf16(aq1, load_ab16(k0p + 32, kb), s0);
            s1 = wmma_bf16(aq0, load_ab16(k1p + 0,  kb), s1);
            s1 = wmma_bf16(aq1, load_ab16(k1p + 32, kb), s1);
        }
        s0 = s0 * SOFTMAX_SCALE;
        s1 = s1 * SOFTMAX_SCALE;

        // ---- online softmax (rows striped across 16-lane half-groups) ----
        v8f corrv;
#pragma unroll
        for (int r = 0; r < 8; ++r) {
            float mx = fmaxf(s0[r], s1[r]);
#pragma unroll
            for (int off = 8; off >= 1; off >>= 1)
                mx = fmaxf(mx, __shfl_xor(mx, off, 32));
            const float mnew = fmaxf(mst[r], mx);
            const float corr = __expf(mst[r] - mnew);
            const float p0 = __expf(s0[r] - mnew);
            const float p1 = __expf(s1[r] - mnew);
            s0[r] = p0; s1[r] = p1;
            float sum = p0 + p1;
#pragma unroll
            for (int off = 8; off >= 1; off >>= 1)
                sum += __shfl_xor(sum, off, 32);
            lst[r] = lst[r] * corr + sum;
            mst[r] = mnew;
            corrv[r] = corr;
        }
#pragma unroll
        for (int t = 0; t < 4; ++t) accO[t] = accO[t] * corrv;

        // ---- P: C-layout -> A-layout via LDS ----
#pragma unroll
        for (int r = 0; r < 8; ++r) {
            const int row = r + 8 * hi;
            pb[row * 32 + lo]      = f2bf_u(s0[r]);
            pb[row * 32 + 16 + lo] = f2bf_u(s1[r]);
        }
        __syncthreads();
        const v16bf pa = load_ab16(pb + lo * 32, kb);
        __syncthreads();

        // ---- ctx += P @ V_block  (Vt rows contiguous in s) ----
#pragma unroll
        for (int t = 0; t < 4; ++t) {
            const v16bf bv = load_ab16(Vp + (size_t)(t * 16 + lo) * SEQ + j, kb);
            accO[t] = wmma_bf16(pa, bv, accO[t]);
        }
    }

    // ---- normalize and write ctx as bf16 [b, s, h*64+d] for the O-GEMM ----
    v8f invv;
#pragma unroll
    for (int r = 0; r < 8; ++r) invv[r] = 1.f / lst[r];
#pragma unroll
    for (int t = 0; t < 4; ++t) {
        const v8f o = accO[t] * invv;
        const int d = t * 16 + lo;
#pragma unroll
        for (int r = 0; r < 8; ++r) {
            const int s = qT * 16 + r + 8 * hi;
            Ctxb[((size_t)bb * SEQ + s) * EMB + h * HD + d] = f2bf_u(o[r]);
        }
    }
}

// =====================================================================
// Kernel 3: output projection, bf16 inputs, wave tile 32x64, fp32 out.
// grid = (EMB/64, B*SEQ/32), block = 32.
// =====================================================================
__global__ void out_gemm_kernel(const unsigned short* __restrict__ Ctxb,
                                const unsigned short* __restrict__ Wob,
                                const float* __restrict__ os,
                                float* __restrict__ out) {
    const int lane = threadIdx.x & 31;
    const int lo = lane & 15;
    const int hi = lane >> 4;
    const int kb = hi * 8;

    const int nT0 = blockIdx.x * 4;
    const int mT0 = blockIdx.y * 2;

    const unsigned short* aRow0 = Ctxb + (size_t)(mT0 * 16 + lo) * EMB;
    const unsigned short* aRow1 = aRow0 + (size_t)16 * EMB;
    const unsigned short* bRow[4];
#pragma unroll
    for (int t = 0; t < 4; ++t)
        bRow[t] = Wob + (size_t)((nT0 + t) * 16 + lo) * EMB;

    v8f acc0[4] = {}, acc1[4] = {};
#pragma unroll 2
    for (int k0 = 0; k0 < EMB; k0 += 32) {
        const v16bf a0 = load_ab16(aRow0 + k0, kb);
        const v16bf a1 = load_ab16(aRow1 + k0, kb);
        __builtin_prefetch(aRow0 + k0 + 64, 0, 1);
#pragma unroll
        for (int t = 0; t < 4; ++t) {
            const v16bf b = load_ab16(bRow[t] + k0, kb);
            acc0[t] = wmma_bf16(a0, b, acc0[t]);
            acc1[t] = wmma_bf16(a1, b, acc1[t]);
        }
    }

#pragma unroll
    for (int t = 0; t < 4; ++t) {
        const int ng = (nT0 + t) * 16 + lo;
        const float scv = os[ng];
#pragma unroll
        for (int m = 0; m < 2; ++m) {
            const v8f av = (m == 0 ? acc0[t] : acc1[t]) * scv;
#pragma unroll
            for (int r = 0; r < 8; ++r) {
                const int mg = (mT0 + m) * 16 + r + 8 * hi;
                out[(size_t)mg * EMB + ng] = av[r];
            }
        }
    }
}

// =====================================================================
extern "C" void kernel_launch(void* const* d_in, const int* in_sizes, int n_in,
                              void* d_out, int out_size, void* d_ws, size_t ws_size,
                              hipStream_t stream) {
    (void)in_sizes; (void)n_in; (void)out_size; (void)ws_size;
    const float* x  = (const float*)d_in[0];
    const float* Wq = (const float*)d_in[1];
    const float* Wk = (const float*)d_in[2];
    const float* Wv = (const float*)d_in[3];
    const float* Wo = (const float*)d_in[4];
    const float* qs = (const float*)d_in[5];
    const float* ks = (const float*)d_in[6];
    const float* vs = (const float*)d_in[7];
    const float* os = (const float*)d_in[8];
    float* out = (float*)d_out;

    // workspace layout (bf16 = ushort):
    //   xb (4M) | Wqb|Wkb|Wvb|Wob (1M each) | Qb|Kb|Vtb|Ctxb (4M each) = 48 MB
    const size_t xe = (size_t)BATCH * SEQ * EMB;   // 4,194,304
    const size_t we = (size_t)EMB * EMB;           // 1,048,576
    unsigned short* xb   = (unsigned short*)d_ws;
    unsigned short* Wqb  = xb + xe;
    unsigned short* Wkb  = Wqb + we;
    unsigned short* Wvb  = Wkb + we;
    unsigned short* Wob  = Wvb + we;
    unsigned short* Qb   = Wob + we;
    unsigned short* Kb   = Qb + xe;
    unsigned short* Vtb  = Kb + xe;
    unsigned short* Ctxb = Vtb + xe;

    cvt_bf16_kernel<<<dim3(512, 5), 256, 0, stream>>>(
        x, Wq, Wk, Wv, Wo, xb, Wqb, Wkb, Wvb, Wob);
    qkv_gemm_kernel<<<dim3(EMB / 64, BATCH * SEQ / 32, 3), 32, 0, stream>>>(
        xb, Wqb, Wkb, Wvb, qs, ks, vs, Qb, Kb, Vtb);
    attn_kernel<<<dim3(SEQ / 16, HEADS, BATCH), 32, 0, stream>>>(
        Qb, Kb, Vtb, Ctxb);
    out_gemm_kernel<<<dim3(EMB / 64, BATCH * SEQ / 32), 32, 0, stream>>>(
        Ctxb, Wob, os, out);
}
